// TidalTransformer_80376017977642
// MI455X (gfx1250) — matrix operand order repaired and checked by reading, not compiled, not run on hardware
//
#include <hip/hip_runtime.h>
#include <math.h>

// ---------------- CDNA5 WMMA types ----------------
typedef __attribute__((ext_vector_type(16))) __bf16 v16bf;
typedef __attribute__((ext_vector_type(8)))  float  v8f;

union BF16Frag { v16bf v; unsigned u[8]; };

__device__ __forceinline__ unsigned short f2bf(float f) {
    union { float f; unsigned u; } x; x.f = f;
    unsigned u = x.u;
    unsigned r = u + 0x7FFFu + ((u >> 16) & 1u);   // round-to-nearest-even
    return (unsigned short)(r >> 16);
}
__device__ __forceinline__ unsigned pack2(unsigned short lo, unsigned short hi) {
    return (unsigned)lo | ((unsigned)hi << 16);
}
// A-fragment (16x32 bf16): lanes 0-15 rows, K = {0..7,16..23} / lanes 16-31 K = {8..15,24..31}
__device__ __forceinline__ int kA(int j, int lane) {
    return ((j < 4) ? 2 * j : 2 * j + 8) + ((lane & 16) ? 8 : 0);
}
// B-fragment (32x16 bf16): lanes 0-15 K = {0..7,16..23} / lanes 16-31 K = {8..15,24..31} (2 per VGPR)
__device__ __forceinline__ int kB(int j, int lane) {
    return ((j < 4) ? 2 * j : 2 * j + 8) + ((lane & 16) ? 8 : 0);
}

// ---------------- GEMM: C = act(A[M,K] x W[K,N] + bias) (+res) ----------------
// block: 256 threads = 8 waves; C tile 32(M) x 128(N); each wave 16x32 (2 accums).
template<bool RELU, bool RES>
__global__ __launch_bounds__(256) void k_gemm_t(const float* __restrict__ A,
                                                const float* __restrict__ W,
                                                const float* __restrict__ bias,
                                                const float* __restrict__ res,
                                                float* __restrict__ C,
                                                int M, int K, int N)
{
    __shared__ unsigned short aS[32][34];    // A tile, [m][k], bf16 bits
    __shared__ unsigned short wS[128][34];   // W tile transposed, [n][k]

    const int tid  = threadIdx.x;
    const int lane = tid & 31;
    const int wv   = tid >> 5;
    const int m0   = blockIdx.x * 32;
    const int n0   = blockIdx.y * 128;
    const int msub = (wv & 1) * 16;
    const int nsub = (wv >> 1) * 32;
    const int nl   = lane & 15;

    v8f acc0, acc1;
#pragma unroll
    for (int j = 0; j < 8; j++) { acc0[j] = 0.f; acc1[j] = 0.f; }

    const int mrow = msub + nl;

    for (int kb = 0; kb < K; kb += 32) {
        {   // stage A tile (32x32) : 4 floats per thread
            int r  = tid >> 3;
            int c4 = (tid & 7) * 4;
            const float* Ap = A + (size_t)(m0 + r) * K + kb + c4;
#pragma unroll
            for (int i = 0; i < 4; i++) aS[r][c4 + i] = f2bf(Ap[i]);
        }
        {   // stage W tile (32x128) transposed into wS[n][k] : 16 floats per thread
            int kk  = tid >> 3;
            int nn0 = (tid & 7) * 16;
            const float* Wp = W + (size_t)(kb + kk) * N + n0 + nn0;
#pragma unroll
            for (int i = 0; i < 16; i++) wS[nn0 + i][kk] = f2bf(Wp[i]);
        }
        __syncthreads();

        BF16Frag a, b0, b1;
#pragma unroll
        for (int j = 0; j < 8; j++) {
            int ka = kA(j, lane), kb2 = kB(j, lane);
            a.u[j]  = *(const unsigned*)&aS[mrow][ka];
            b0.u[j] = *(const unsigned*)&wS[nsub + nl][kb2];
            b1.u[j] = *(const unsigned*)&wS[nsub + 16 + nl][kb2];
        }
        acc0 = __builtin_amdgcn_wmma_f32_16x16x32_bf16(false, a.v, false, b0.v, (short)0, acc0, false, false);
        acc1 = __builtin_amdgcn_wmma_f32_16x16x32_bf16(false, a.v, false, b1.v, (short)0, acc1, false, false);
        __syncthreads();
    }

    const int col0 = n0 + nsub + nl;
    const int col1 = col0 + 16;
    const float bv0 = bias[col0];
    const float bv1 = bias[col1];
#pragma unroll
    for (int j = 0; j < 8; j++) {
        int row = m0 + msub + j + ((lane & 16) ? 8 : 0);
        float v0 = acc0[j] + bv0;
        float v1 = acc1[j] + bv1;
        if (RES) {
            v0 += res[(size_t)row * N + col0];
            v1 += res[(size_t)row * N + col1];
        }
        if (RELU) { v0 = fmaxf(v0, 0.f); v1 = fmaxf(v1, 0.f); }
        C[(size_t)row * N + col0] = v0;
        C[(size_t)row * N + col1] = v1;
    }
}

// ---------------- Flash attention: one wave per (32-query block, head) ----------------
// Q,K,V,O are [seq,256] f32; head h occupies cols h*32..h*32+31. DH=32 == WMMA K.
// Two 16-row query tiles per wave share each staged 32-key K/V block (8 WMMA / block).
__global__ __launch_bounds__(32) void k_attn(const float* __restrict__ Q,
                                             const float* __restrict__ Km,
                                             const float* __restrict__ Vm,
                                             float* __restrict__ O,
                                             int sk, int causal)
{
    __shared__ unsigned short kS[32][34];   // [key][dim]
    __shared__ unsigned short vS[32][34];   // [dim][key]
    __shared__ unsigned short pS[32][34];   // [m (2 tiles)][key]

    const int lane = threadIdx.x;
    const int qb   = blockIdx.x * 32;
    const int hoff = blockIdx.y * 32;
    const int nl   = lane & 15;
    const int hi   = (lane & 16) ? 8 : 0;
    const float scale = 0.17677669529663687f;   // 1/sqrt(32)

    BF16Frag qf[2];
#pragma unroll
    for (int t = 0; t < 2; t++) {
        const float* Qp = Q + (size_t)(qb + t * 16 + nl) * 256 + hoff;
#pragma unroll
        for (int j = 0; j < 8; j++) {
            int k2 = kA(j, lane);
            qf[t].u[j] = pack2(f2bf(Qp[k2]), f2bf(Qp[k2 + 1]));
        }
    }

    float mi[2][8], li[2][8];
    v8f o00, o01, o10, o11;   // o[tile][dim-chunk]
#pragma unroll
    for (int j = 0; j < 8; j++) {
        mi[0][j] = -INFINITY; mi[1][j] = -INFINITY;
        li[0][j] = 0.f; li[1][j] = 0.f;
        o00[j] = 0.f; o01[j] = 0.f; o10[j] = 0.f; o11[j] = 0.f;
    }

    const int nkb = causal ? (qb + 32 + 31) / 32 : (sk + 31) / 32;

    for (int b = 0; b < nkb; b++) {
        const int k0 = b * 32;
        {   // stage K and V blocks: lane handles key row (k0+lane), coalesced reads
            const float* Kp = Km + (size_t)(k0 + lane) * 256 + hoff;
            const float* Vp = Vm + (size_t)(k0 + lane) * 256 + hoff;
#pragma unroll
            for (int d = 0; d < 32; d++) {
                kS[lane][d] = f2bf(Kp[d]);
                vS[d][lane] = f2bf(Vp[d]);
            }
        }
        __syncthreads();

        BF16Frag b0, b1;
#pragma unroll
        for (int j = 0; j < 8; j++) {
            int k2 = kB(j, lane);
            b0.u[j] = *(const unsigned*)&kS[nl][k2];
            b1.u[j] = *(const unsigned*)&kS[16 + nl][k2];
        }

#pragma unroll
        for (int t = 0; t < 2; t++) {
            v8f s0, s1;
#pragma unroll
            for (int j = 0; j < 8; j++) { s0[j] = 0.f; s1[j] = 0.f; }
            s0 = __builtin_amdgcn_wmma_f32_16x16x32_bf16(false, qf[t].v, false, b0.v, (short)0, s0, false, false);
            s1 = __builtin_amdgcn_wmma_f32_16x16x32_bf16(false, qf[t].v, false, b1.v, (short)0, s1, false, false);

            const int rowg = qb + t * 16 + hi;
            float mnew[8];
#pragma unroll
            for (int j = 0; j < 8; j++) {
                float a0 = s0[j] * scale, a1 = s1[j] * scale;
                if (causal) {
                    int r = rowg + j;
                    if (k0 + nl > r)      a0 = -1e9f;
                    if (k0 + 16 + nl > r) a1 = -1e9f;
                }
                s0[j] = a0; s1[j] = a1;
                float ml = fmaxf(a0, a1);
                for (int d = 1; d < 16; d <<= 1) ml = fmaxf(ml, __shfl_xor(ml, d, 32));
                mnew[j] = fmaxf(mi[t][j], ml);
            }
#pragma unroll
            for (int j = 0; j < 8; j++) {
                float p0 = __expf(s0[j] - mnew[j]);
                float p1 = __expf(s1[j] - mnew[j]);
                float ls = p0 + p1;
                for (int d = 1; d < 16; d <<= 1) ls += __shfl_xor(ls, d, 32);
                float corr = __expf(mi[t][j] - mnew[j]);
                li[t][j] = li[t][j] * corr + ls;
                mi[t][j] = mnew[j];
                if (t == 0) { o00[j] *= corr; o01[j] *= corr; }
                else        { o10[j] *= corr; o11[j] *= corr; }
                int mr = t * 16 + j + hi;
                pS[mr][nl]      = f2bf(p0);
                pS[mr][16 + nl] = f2bf(p1);
            }
        }
        __syncthreads();

        BF16Frag bv0, bv1, pa0, pa1;
#pragma unroll
        for (int j = 0; j < 8; j++) {
            int ka = kA(j, lane), k2 = kB(j, lane);
            bv0.u[j] = *(const unsigned*)&vS[nl][k2];
            bv1.u[j] = *(const unsigned*)&vS[16 + nl][k2];
            pa0.u[j] = *(const unsigned*)&pS[nl][ka];
            pa1.u[j] = *(const unsigned*)&pS[16 + nl][ka];
        }
        o00 = __builtin_amdgcn_wmma_f32_16x16x32_bf16(false, pa0.v, false, bv0.v, (short)0, o00, false, false);
        o01 = __builtin_amdgcn_wmma_f32_16x16x32_bf16(false, pa0.v, false, bv1.v, (short)0, o01, false, false);
        o10 = __builtin_amdgcn_wmma_f32_16x16x32_bf16(false, pa1.v, false, bv0.v, (short)0, o10, false, false);
        o11 = __builtin_amdgcn_wmma_f32_16x16x32_bf16(false, pa1.v, false, bv1.v, (short)0, o11, false, false);
        __syncthreads();
    }

#pragma unroll
    for (int j = 0; j < 8; j++) {
        float inv0 = 1.0f / li[0][j];
        float inv1 = 1.0f / li[1][j];
        int row0 = qb + j + hi;
        int row1 = row0 + 16;
        O[(size_t)row0 * 256 + hoff + nl]      = o00[j] * inv0;
        O[(size_t)row0 * 256 + hoff + 16 + nl] = o01[j] * inv0;
        O[(size_t)row1 * 256 + hoff + nl]      = o10[j] * inv1;
        O[(size_t)row1 * 256 + hoff + 16 + nl] = o11[j] * inv1;
    }
}

// ---------------- embedding + sinusoidal PE ----------------
__global__ __launch_bounds__(256) void k_embed(float* __restrict__ x,
                                               const float* __restrict__ src,
                                               const float* __restrict__ inW,
                                               const float* __restrict__ inB)
{
    int t = blockIdx.x, d = threadIdx.x;
    float div = __expf((float)(d & ~1) * (-9.210340371976184f / 256.f));
    float ang = (float)t * div;
    float pe  = (d & 1) ? __cosf(ang) : __sinf(ang);
    x[(size_t)t * 256 + d] = src[t] * inW[d] + inB[d] + pe;
}

// ---------------- LayerNorm (wave per row, D=256) ----------------
__global__ __launch_bounds__(32) void k_ln(float* __restrict__ out,
                                           const float* __restrict__ in,
                                           const float* __restrict__ g,
                                           const float* __restrict__ b)
{
    int r = blockIdx.x, lane = threadIdx.x;
    const float* x = in + (size_t)r * 256;
    float v[8], s = 0.f;
#pragma unroll
    for (int i = 0; i < 8; i++) { v[i] = x[lane + 32 * i]; s += v[i]; }
    for (int d = 1; d < 32; d <<= 1) s += __shfl_xor(s, d, 32);
    float mean = s * (1.f / 256.f);
    float vs = 0.f;
#pragma unroll
    for (int i = 0; i < 8; i++) { float t = v[i] - mean; vs += t * t; }
    for (int d = 1; d < 32; d <<= 1) vs += __shfl_xor(vs, d, 32);
    float rstd = rsqrtf(vs * (1.f / 256.f) + 1e-5f);
#pragma unroll
    for (int i = 0; i < 8; i++) {
        int c = lane + 32 * i;
        out[(size_t)r * 256 + c] = (v[i] - mean) * rstd * g[c] + b[c];
    }
}

// ---------------- output projection: [T,256] x [256,1] + b ----------------
__global__ __launch_bounds__(32) void k_outproj(float* __restrict__ out,
                                                const float* __restrict__ y,
                                                const float* __restrict__ W,
                                                const float* __restrict__ b)
{
    int r = blockIdx.x, lane = threadIdx.x;
    float s = 0.f;
#pragma unroll
    for (int i = 0; i < 8; i++) { int c = lane + 32 * i; s += y[(size_t)r * 256 + c] * W[c]; }
    for (int d = 1; d < 32; d <<= 1) s += __shfl_xor(s, d, 32);
    if (lane == 0) out[r] = s + b[0];
}

// ---------------- host schedule ----------------
static inline void gemm(hipStream_t s, const float* A, const float* W, const float* bias,
                        const float* res, float* C, int M, int K, int N, int relu)
{
    dim3 g(M / 32, N / 128), blk(256);
    if (res)
        k_gemm_t<false, true><<<g, blk, 0, s>>>(A, W, bias, res, C, M, K, N);
    else if (relu)
        k_gemm_t<true, false><<<g, blk, 0, s>>>(A, W, bias, res, C, M, K, N);
    else
        k_gemm_t<false, false><<<g, blk, 0, s>>>(A, W, bias, res, C, M, K, N);
}

extern "C" void kernel_launch(void* const* d_in, const int* in_sizes, int n_in,
                              void* d_out, int out_size, void* d_ws, size_t ws_size,
                              hipStream_t stream)
{
    (void)in_sizes; (void)n_in; (void)out_size;
    auto F = [&](int i) { return (const float*)d_in[i]; };

    const int SL = 4320, TL = 1440, D = 256, FFN = 1024;
    const size_t need = (size_t)(7 * SL * D + SL * FFN + TL * D) * sizeof(float);
    if (ws_size < need) return;

    float* ws  = (float*)d_ws;
    float* x   = ws;                     // [SL,D] encoder activations
    float* tmp = x   + (size_t)SL * D;
    float* q   = tmp + (size_t)SL * D;
    float* kk  = q   + (size_t)SL * D;
    float* vv  = kk  + (size_t)SL * D;
    float* ao  = vv  + (size_t)SL * D;
    float* mem = ao  + (size_t)SL * D;
    float* hid = mem + (size_t)SL * D;   // [SL,FF]
    float* y   = hid + (size_t)SL * FFN; // [TL,D]

    // pytree order: dec(3x26), dec_ln(b,g), enc(6x16), enc_ln(b,g), in_W,in_b,out_W,out_b, src, tgt
    const int IN_W = 178, IN_B = 179, OUT_W = 180, OUT_B = 181, SRC = 182, TGT = 183;

    // ---- encoder ----
    k_embed<<<SL, 256, 0, stream>>>(x, F(SRC), F(IN_W), F(IN_B));
    for (int l = 0; l < 6; l++) {
        int E = 80 + 16 * l; // ff.W1,W2,b1,b2 | ln1.b,g | ln2.b,g | sa.Wk,Wo,Wq,Wv,bk,bo,bq,bv
        gemm(stream, x, F(E + 10), F(E + 14), nullptr, q,  SL, D, D, 0);  // Q
        gemm(stream, x, F(E + 8),  F(E + 12), nullptr, kk, SL, D, D, 0);  // K
        gemm(stream, x, F(E + 11), F(E + 15), nullptr, vv, SL, D, D, 0);  // V
        k_attn<<<dim3(SL / 32, 8), 32, 0, stream>>>(q, kk, vv, ao, SL, 0);
        gemm(stream, ao, F(E + 9), F(E + 13), x, tmp, SL, D, D, 0);       // Wo + residual
        k_ln<<<SL, 32, 0, stream>>>(x, tmp, F(E + 5), F(E + 4));          // ln1
        gemm(stream, x,   F(E + 0), F(E + 2), nullptr, hid, SL, D,  FFN, 1); // W1+relu
        gemm(stream, hid, F(E + 1), F(E + 3), x, tmp,       SL, FFN, D,  0); // W2 + residual
        k_ln<<<SL, 32, 0, stream>>>(x, tmp, F(E + 7), F(E + 6));          // ln2
    }
    k_ln<<<SL, 32, 0, stream>>>(mem, x, F(177), F(176));                  // enc_ln

    // ---- decoder ----
    k_embed<<<TL, 256, 0, stream>>>(y, F(TGT), F(IN_W), F(IN_B));
    for (int l = 0; l < 3; l++) {
        int Dd = 26 * l; // ca.Wk,Wo,Wq,Wv,bk,bo,bq,bv | ff.W1,W2,b1,b2 | ln1 b,g | ln2 b,g | ln3 b,g | sa.(8)
        // causal self-attention
        gemm(stream, y, F(Dd + 20), F(Dd + 24), nullptr, q,  TL, D, D, 0);
        gemm(stream, y, F(Dd + 18), F(Dd + 22), nullptr, kk, TL, D, D, 0);
        gemm(stream, y, F(Dd + 21), F(Dd + 25), nullptr, vv, TL, D, D, 0);
        k_attn<<<dim3(TL / 32, 8), 32, 0, stream>>>(q, kk, vv, ao, TL, 1);
        gemm(stream, ao, F(Dd + 19), F(Dd + 23), y, tmp, TL, D, D, 0);
        k_ln<<<TL, 32, 0, stream>>>(y, tmp, F(Dd + 13), F(Dd + 12));
        // cross attention (K/V from mem with this layer's weights)
        gemm(stream, y,   F(Dd + 2), F(Dd + 6), nullptr, q,  TL, D, D, 0);
        gemm(stream, mem, F(Dd + 0), F(Dd + 4), nullptr, kk, SL, D, D, 0);
        gemm(stream, mem, F(Dd + 3), F(Dd + 7), nullptr, vv, SL, D, D, 0);
        k_attn<<<dim3(TL / 32, 8), 32, 0, stream>>>(q, kk, vv, ao, SL, 0);
        gemm(stream, ao, F(Dd + 1), F(Dd + 5), y, tmp, TL, D, D, 0);
        k_ln<<<TL, 32, 0, stream>>>(y, tmp, F(Dd + 15), F(Dd + 14));
        // FFN
        gemm(stream, y,   F(Dd + 8), F(Dd + 10), nullptr, hid, TL, D,  FFN, 1);
        gemm(stream, hid, F(Dd + 9), F(Dd + 11), y, tmp,       TL, FFN, D,  0);
        k_ln<<<TL, 32, 0, stream>>>(y, tmp, F(Dd + 17), F(Dd + 16));
    }
    k_ln<<<TL, 32, 0, stream>>>(y, y, F(79), F(78));                      // dec_ln (in-place safe)
    k_outproj<<<TL, 32, 0, stream>>>((float*)d_out, y, F(OUT_W), F(OUT_B));
}
